// OIMLoss_13116830122679
// MI455X (gfx1250) — matrix operand order repaired and checked by reading, not compiled, not run on hardware
//
#include <hip/hip_runtime.h>
#include <hip/hip_bf16.h>

#define NROWS      8192
#define DIMK       256
#define NPIDS      5532
#define NCQ        5000
#define LTOT       (NPIDS + NCQ)      // 10532
#define LPAD       10544              // 659 * 16, zero padded
#define IGNORE_IDX 5554
#define OIMS       30.0f
#define MTILE      128
#define NTILES     (LPAD / 16)        // 659
#define TILE_BYTES (16 * DIMK * 2)    // 8192 bytes per plane per tile

typedef __attribute__((ext_vector_type(16))) __bf16 v16bf;
typedef __attribute__((ext_vector_type(8)))  float  v8f;

union FragU { v16bf v; uint4 q[2]; };

#if defined(__HIP_DEVICE_COMPILE__) && __has_builtin(__builtin_amdgcn_global_load_async_to_lds_b128)
#define HAVE_ASYNC 1
typedef int v4i_gcc __attribute__((vector_size(16)));
typedef __attribute__((address_space(1))) v4i_gcc GVec;   // prints as "__device__" in diags
typedef __attribute__((address_space(3))) v4i_gcc LVec;   // prints as "__shared__" in diags
#else
#define HAVE_ASYNC 0
#endif

__device__ __forceinline__ void copy16_g2l(const char* g, char* l) {
#if HAVE_ASYNC
  __builtin_amdgcn_global_load_async_to_lds_b128((GVec*)g, (LVec*)l, 0, 0);
#else
  *(uint4*)l = *(const uint4*)g;
#endif
}

// ---------------------------------------------------------------------------
// Pre-conversion: bank[col][k] * reliability[col] * 30 -> bf16 hi/lo planes,
// zero-padded to LPAD columns. One-shot; removes all conversion VALU from the
// 64-block hot loop.
// ---------------------------------------------------------------------------
__global__ __launch_bounds__(256) void oim_precvt_kernel(
    const float* __restrict__ lut, const float* __restrict__ cq,
    const float* __restrict__ rel,
    __bf16* __restrict__ hi, __bf16* __restrict__ lo)
{
  const int c   = blockIdx.x * 256 + threadIdx.x;   // one chunk = 4 floats
  const int row = c >> 6;
  const int kc  = (c & 63) << 2;
  if (row >= LPAD) return;

  float4 f = make_float4(0.f, 0.f, 0.f, 0.f);
  float  r = 0.f;
  if (row < LTOT) {
    const float* src = (row < NPIDS) ? (lut + (size_t)row * DIMK + kc)
                                     : (cq  + (size_t)(row - NPIDS) * DIMK + kc);
    f = *(const float4*)src;
    r = rel[row] * OIMS;
  }
  float x[4] = {f.x * r, f.y * r, f.z * r, f.w * r};
  const size_t o = (size_t)row * DIMK + kc;
  #pragma unroll
  for (int t = 0; t < 4; ++t) {
    __bf16 h = (__bf16)x[t];
    hi[o + t] = h;
    lo[o + t] = (__bf16)(x[t] - (float)h);
  }
}

// ---------------------------------------------------------------------------
// Fast main kernel: pre-converted bank, async triple-buffered B staging.
// ---------------------------------------------------------------------------
__global__ __launch_bounds__(256) void oim_main_fast_kernel(
    const float* __restrict__ inputs,
    const int*   __restrict__ label,
    const __bf16* __restrict__ bhi,
    const __bf16* __restrict__ blo,
    float* __restrict__ parts)
{
  __shared__ __align__(16) __bf16 sB[3][2][16][DIMK];  // [buf][plane][col][k], 48 KB
  __shared__ float sRedS[16];
  __shared__ float sRedC[16];

  const int tid  = threadIdx.x;
  const int lane = tid & 31;
  const int wave = tid >> 5;
  const int hg   = lane >> 4;
  const int lr   = lane & 15;

  const int rowbase = blockIdx.x * MTILE + wave * 16;

  // ---- A fragments (16 rows x K=256), contiguous k = 16*hg + e layout, bf16 hi/lo split, in regs
  v16bf a_hi[8], a_lo[8];
  {
    const float* arow = inputs + (size_t)(rowbase + lr) * DIMK;
    #pragma unroll
    for (int kb = 0; kb < 8; ++kb) {
      const int k0 = kb * 32 + 16 * hg;
      float4 c0 = *(const float4*)(arow + k0);
      float4 c1 = *(const float4*)(arow + k0 + 4);
      float4 c2 = *(const float4*)(arow + k0 + 8);
      float4 c3 = *(const float4*)(arow + k0 + 12);
      float xs[16] = {c0.x,c0.y,c0.z,c0.w, c1.x,c1.y,c1.z,c1.w,
                      c2.x,c2.y,c2.z,c2.w, c3.x,c3.y,c3.z,c3.w};
      #pragma unroll
      for (int e = 0; e < 16; ++e) {
        __bf16 h = (__bf16)xs[e];
        a_hi[kb][e] = h;
        a_lo[kb][e] = (__bf16)(xs[e] - (float)h);
      }
    }
  }

  int lbl[8];
  #pragma unroll
  for (int i = 0; i < 8; ++i) lbl[i] = label[rowbase + 8 * hg + i];

  float lmax[8], lsum[8], tgt[8];
  #pragma unroll
  for (int i = 0; i < 8; ++i) { lmax[i] = -1e30f; lsum[i] = 0.0f; tgt[i] = -1e30f; }

  // 16KB tile copy: 1024 x 16B chunks over 256 threads -> 4 chunks/thread
  auto stage = [&](int buf, int nt) {
    const char* ghi = (const char*)bhi + (size_t)nt * TILE_BYTES;
    const char* glo = (const char*)blo + (size_t)nt * TILE_BYTES;
    char* lbase = (char*)&sB[buf][0][0][0];
    #pragma unroll
    for (int j = 0; j < 4; ++j) {
      const int c     = tid + 256 * j;        // 0..1023
      const int plane = c >> 9;               // 0 = hi, 1 = lo
      const int off   = (c & 511) << 4;       // byte offset within plane
      copy16_g2l((plane ? glo : ghi) + off, lbase + plane * TILE_BYTES + off);
    }
  };

  // prologue: prefetch tiles 0 and 1
  stage(0, 0);
  stage(1, 1);

  for (int nt = 0; nt < NTILES; ++nt) {
#if HAVE_ASYNC
    // tile nt's 4 async chunks retire in order; tile nt+1's may stay in flight
    if (nt == NTILES - 1) asm volatile("s_wait_asynccnt 0" ::: "memory");
    else                  asm volatile("s_wait_asynccnt 4" ::: "memory");
#endif
    __syncthreads();                 // tile nt staged by all waves; buf (nt+2)%3 free
    if (nt + 2 < NTILES) stage((nt + 2) % 3, nt + 2);

    const int cur = nt % 3;
    v8f acc = {0.f, 0.f, 0.f, 0.f, 0.f, 0.f, 0.f, 0.f};
    #pragma unroll
    for (int kb = 0; kb < 8; ++kb) {
      FragU bh, bl;
      const __bf16* ph = &sB[cur][0][lr][kb * 32 + 16 * hg];
      const __bf16* pl = &sB[cur][1][lr][kb * 32 + 16 * hg];
      bh.q[0] = *(const uint4*)(ph);
      bh.q[1] = *(const uint4*)(ph + 8);
      bl.q[0] = *(const uint4*)(pl);
      bl.q[1] = *(const uint4*)(pl + 8);
      acc = __builtin_amdgcn_wmma_f32_16x16x32_bf16(false, a_hi[kb], false, bh.v, (short)0, acc, false, false);
      acc = __builtin_amdgcn_wmma_f32_16x16x32_bf16(false, a_hi[kb], false, bl.v, (short)0, acc, false, false);
      acc = __builtin_amdgcn_wmma_f32_16x16x32_bf16(false, a_lo[kb], false, bh.v, (short)0, acc, false, false);
    }

    // scale already baked into B; online log-sum-exp per lane column
    const int gcol = nt * 16 + lr;
    #pragma unroll
    for (int i = 0; i < 8; ++i) {
      float vv = (gcol < LTOT) ? acc[i] : -1e30f;
      if (gcol == lbl[i]) tgt[i] = vv;
      float nm = fmaxf(lmax[i], vv);
      lsum[i] = lsum[i] * __expf(lmax[i] - nm) + __expf(vv - nm);
      lmax[i] = nm;
    }
  }

  // merge per-row stats across 16 lanes of each half-group
  #pragma unroll
  for (int off = 1; off < 16; off <<= 1) {
    #pragma unroll
    for (int i = 0; i < 8; ++i) {
      float om = __shfl_xor(lmax[i], off, 32);
      float os = __shfl_xor(lsum[i], off, 32);
      float ot = __shfl_xor(tgt[i],  off, 32);
      float nm = fmaxf(lmax[i], om);
      lsum[i] = lsum[i] * __expf(lmax[i] - nm) + os * __expf(om - nm);
      lmax[i] = nm;
      tgt[i]  = fmaxf(tgt[i], ot);
    }
  }

  if (lr == 0) {
    float s = 0.f, c = 0.f;
    #pragma unroll
    for (int i = 0; i < 8; ++i) {
      if (lbl[i] != IGNORE_IDX) {
        s += (lmax[i] + __logf(lsum[i])) - tgt[i];
        c += 1.0f;
      }
    }
    sRedS[wave * 2 + hg] = s;
    sRedC[wave * 2 + hg] = c;
  }
  __syncthreads();
  if (tid == 0) {
    float s = 0.f, c = 0.f;
    for (int i = 0; i < 16; ++i) { s += sRedS[i]; c += sRedC[i]; }
    parts[blockIdx.x]      = s;
    parts[64 + blockIdx.x] = c;
  }
}

// ---------------------------------------------------------------------------
// Legacy self-contained kernel (512 B workspace) — fallback if ws too small.
// ---------------------------------------------------------------------------
__global__ __launch_bounds__(256) void oim_main_legacy_kernel(
    const float* __restrict__ inputs,
    const int*   __restrict__ label,
    const float* __restrict__ lut,
    const float* __restrict__ cq,
    const float* __restrict__ rel,
    float* __restrict__ parts)
{
  __shared__ __align__(16) __bf16 sBhi[16][DIMK];
  __shared__ __align__(16) __bf16 sBlo[16][DIMK];
  __shared__ float sRel[LTOT];
  __shared__ float sRedS[16];
  __shared__ float sRedC[16];

  const int tid  = threadIdx.x;
  const int lane = tid & 31;
  const int wave = tid >> 5;
  const int hg   = lane >> 4;
  const int lr   = lane & 15;

  for (int i = tid; i < LTOT; i += 256) sRel[i] = rel[i] * OIMS;

  const int rowbase = blockIdx.x * MTILE + wave * 16;

  v16bf a_hi[8], a_lo[8];
  {
    const float* arow = inputs + (size_t)(rowbase + lr) * DIMK;
    #pragma unroll
    for (int kb = 0; kb < 8; ++kb) {
      const int k0 = kb * 32 + 16 * hg;
      float4 c0 = *(const float4*)(arow + k0);
      float4 c1 = *(const float4*)(arow + k0 + 4);
      float4 c2 = *(const float4*)(arow + k0 + 8);
      float4 c3 = *(const float4*)(arow + k0 + 12);
      float xs[16] = {c0.x,c0.y,c0.z,c0.w, c1.x,c1.y,c1.z,c1.w,
                      c2.x,c2.y,c2.z,c2.w, c3.x,c3.y,c3.z,c3.w};
      #pragma unroll
      for (int e = 0; e < 16; ++e) {
        __bf16 h = (__bf16)xs[e];
        a_hi[kb][e] = h;
        a_lo[kb][e] = (__bf16)(xs[e] - (float)h);
      }
    }
  }

  int lbl[8];
  #pragma unroll
  for (int i = 0; i < 8; ++i) lbl[i] = label[rowbase + 8 * hg + i];

  float lmax[8], lsum[8], tgt[8];
  #pragma unroll
  for (int i = 0; i < 8; ++i) { lmax[i] = -1e30f; lsum[i] = 0.0f; tgt[i] = -1e30f; }

  __syncthreads();

  for (int nt = 0; nt < NTILES; ++nt) {
    {
      const int col  = tid >> 4;
      const int ks   = (tid & 15) << 4;
      const int gcol = nt * 16 + col;
      const bool bv  = gcol < LTOT;
      const float* src = lut;
      if (bv) src = (gcol < NPIDS) ? (lut + (size_t)gcol * DIMK + ks)
                                   : (cq  + (size_t)(gcol - NPIDS) * DIMK + ks);
      #pragma unroll
      for (int j = 0; j < 16; j += 4) {
        float4 f = bv ? *(const float4*)(src + j) : make_float4(0.f, 0.f, 0.f, 0.f);
        float fv[4] = {f.x, f.y, f.z, f.w};
        #pragma unroll
        for (int t = 0; t < 4; ++t) {
          __bf16 h = (__bf16)fv[t];
          sBhi[col][ks + j + t] = h;
          sBlo[col][ks + j + t] = (__bf16)(fv[t] - (float)h);
        }
      }
    }
    __syncthreads();

    v8f acc = {0.f, 0.f, 0.f, 0.f, 0.f, 0.f, 0.f, 0.f};
    #pragma unroll
    for (int kb = 0; kb < 8; ++kb) {
      FragU bh, bl;
      const __bf16* ph = &sBhi[lr][kb * 32 + 16 * hg];
      const __bf16* pl = &sBlo[lr][kb * 32 + 16 * hg];
      bh.q[0] = *(const uint4*)(ph);
      bh.q[1] = *(const uint4*)(ph + 8);
      bl.q[0] = *(const uint4*)(pl);
      bl.q[1] = *(const uint4*)(pl + 8);
      acc = __builtin_amdgcn_wmma_f32_16x16x32_bf16(false, a_hi[kb], false, bh.v, (short)0, acc, false, false);
      acc = __builtin_amdgcn_wmma_f32_16x16x32_bf16(false, a_hi[kb], false, bl.v, (short)0, acc, false, false);
      acc = __builtin_amdgcn_wmma_f32_16x16x32_bf16(false, a_lo[kb], false, bh.v, (short)0, acc, false, false);
    }

    const int  gcol = nt * 16 + lr;
    const bool cval = gcol < LTOT;
    const float r30 = cval ? sRel[gcol] : 0.0f;
    #pragma unroll
    for (int i = 0; i < 8; ++i) {
      float vv = cval ? acc[i] * r30 : -1e30f;
      if (gcol == lbl[i]) tgt[i] = vv;
      float nm = fmaxf(lmax[i], vv);
      lsum[i] = lsum[i] * __expf(lmax[i] - nm) + __expf(vv - nm);
      lmax[i] = nm;
    }
    __syncthreads();
  }

  #pragma unroll
  for (int off = 1; off < 16; off <<= 1) {
    #pragma unroll
    for (int i = 0; i < 8; ++i) {
      float om = __shfl_xor(lmax[i], off, 32);
      float os = __shfl_xor(lsum[i], off, 32);
      float ot = __shfl_xor(tgt[i],  off, 32);
      float nm = fmaxf(lmax[i], om);
      lsum[i] = lsum[i] * __expf(lmax[i] - nm) + os * __expf(om - nm);
      lmax[i] = nm;
      tgt[i]  = fmaxf(tgt[i], ot);
    }
  }

  if (lr == 0) {
    float s = 0.f, c = 0.f;
    #pragma unroll
    for (int i = 0; i < 8; ++i) {
      if (lbl[i] != IGNORE_IDX) {
        s += (lmax[i] + __logf(lsum[i])) - tgt[i];
        c += 1.0f;
      }
    }
    sRedS[wave * 2 + hg] = s;
    sRedC[wave * 2 + hg] = c;
  }
  __syncthreads();
  if (tid == 0) {
    float s = 0.f, c = 0.f;
    for (int i = 0; i < 16; ++i) { s += sRedS[i]; c += sRedC[i]; }
    parts[blockIdx.x]      = s;
    parts[64 + blockIdx.x] = c;
  }
}

__global__ void oim_finalize_kernel(const float* __restrict__ parts, float* __restrict__ out)
{
  if (threadIdx.x == 0 && blockIdx.x == 0) {
    float s = 0.f, c = 0.f;
    for (int i = 0; i < 64; ++i) { s += parts[i]; c += parts[64 + i]; }  // fixed order
    out[0] = s / fmaxf(c, 1.0f);
  }
}

extern "C" void kernel_launch(void* const* d_in, const int* in_sizes, int n_in,
                              void* d_out, int out_size, void* d_ws, size_t ws_size,
                              hipStream_t stream) {
  (void)in_sizes; (void)n_in; (void)out_size;
  const float* inputs = (const float*)d_in[0];
  const int*   label  = (const int*)  d_in[1];
  // d_in[2] = ious: unused by the reference computation
  const float* lut    = (const float*)d_in[3];
  const float* cq     = (const float*)d_in[4];
  const float* rel    = (const float*)d_in[5];
  float*       out    = (float*)d_out;

  const size_t plane_bytes = (size_t)LPAD * DIMK * 2;           // 5,398,528
  const size_t need        = 2 * plane_bytes + 128 * sizeof(float);

  if (ws_size >= need) {
    __bf16* hi    = (__bf16*)d_ws;
    __bf16* lo    = (__bf16*)((char*)d_ws + plane_bytes);
    float*  parts = (float*)((char*)d_ws + 2 * plane_bytes);
    oim_precvt_kernel<<<(LPAD * 64) / 256, 256, 0, stream>>>(lut, cq, rel, hi, lo);
    oim_main_fast_kernel<<<NROWS / MTILE, 256, 0, stream>>>(inputs, label, hi, lo, parts);
    oim_finalize_kernel<<<1, 32, 0, stream>>>(parts, out);
  } else {
    float* parts = (float*)d_ws;
    oim_main_legacy_kernel<<<NROWS / MTILE, 256, 0, stream>>>(inputs, label, lut, cq, rel, parts);
    oim_finalize_kernel<<<1, 32, 0, stream>>>(parts, out);
  }
}
